// Block_781684048597
// MI455X (gfx1250) — matrix operand order repaired and checked by reading, not compiled, not run on hardware
//
#include <hip/hip_runtime.h>
#include <hip/hip_bf16.h>
#include <math.h>

typedef __bf16 bf16;
typedef __attribute__((ext_vector_type(16))) __bf16 v16bf;
typedef __attribute__((ext_vector_type(8)))  __bf16 v8bf;
typedef __attribute__((ext_vector_type(8)))  float  v8f;

#define DEVINL __device__ __forceinline__

// ---------------- WMMA fragment helpers (gfx1250 wave32 layouts) ----------------
// A 16x32 bf16: lane -> M = lane%16; half = lane/16.
//   elems 0..7  = K[kBase + half*8 + 0..7]
//   elems 8..15 = K[kBase + 16 + half*8 + 0..7]
DEVINL v16bf load_a_at(const bf16* laneRow) {
  v8bf lo = *(const v8bf*)laneRow;
  v8bf hi = *(const v8bf*)(laneRow + 16);
  return __builtin_shufflevector(lo, hi, 0,1,2,3,4,5,6,7,8,9,10,11,12,13,14,15);
}

// B 32x16 bf16, stored transposed as Bt[N][K]: lane -> N = lane%16; half = lane/16.
//   elems 0..7  = K[kBase + half*8 + 0..7]
//   elems 8..15 = K[kBase + 16 + half*8 + 0..7]
DEVINL v16bf load_b_at(const bf16* laneRow) {
  v8bf lo = *(const v8bf*)laneRow;
  v8bf hi = *(const v8bf*)(laneRow + 16);
  return __builtin_shufflevector(lo, hi, 0,1,2,3,4,5,6,7,8,9,10,11,12,13,14,15);
}

DEVINL v16bf load_a_frag(const bf16* A, int lda, int mBase, int kBase, int lane) {
  const int half = lane >> 4;
  const int m    = mBase + (lane & 15);
  return load_a_at(A + (size_t)m * lda + kBase + half * 8);
}

DEVINL v16bf load_b_frag(const bf16* Bt, int ldb, int nBase, int kBase, int lane) {
  const int half = lane >> 4;
  const int n    = nBase + (lane & 15);
  return load_b_at(Bt + (size_t)n * ldb + kBase + half * 8);
}

DEVINL v8f wmma_bf16(v16bf a, v16bf b, v8f c) {
  return __builtin_amdgcn_wmma_f32_16x16x32_bf16(false, a, false, b, (short)0, c, false, false);
}

DEVINL float redmax16(float v) {
  #pragma unroll
  for (int off = 1; off < 16; off <<= 1) v = fmaxf(v, __shfl_xor(v, off, 32));
  return v;
}
DEVINL float redsum16(float v) {
  #pragma unroll
  for (int off = 1; off < 16; off <<= 1) v += __shfl_xor(v, off, 32);
  return v;
}
DEVINL float redsum32(float v) {
  #pragma unroll
  for (int off = 1; off < 32; off <<= 1) v += __shfl_xor(v, off, 32);
  return v;
}

// ------------- weight transpose + cast: W[K][N] f32 -> Wt[N][K] bf16 (LDS tiled) -------------
__global__ __launch_bounds__(256)
void transpose_bf16_kernel(const float* __restrict__ W, bf16* __restrict__ Wt, int K, int N) {
  __shared__ bf16 tile[32][33];
  const int kt = blockIdx.y * 32, nt = blockIdx.x * 32;
  const int tx = threadIdx.x, ty = threadIdx.y;  // 32 x 8
  #pragma unroll
  for (int r = 0; r < 32; r += 8)
    tile[ty + r][tx] = (bf16)W[(size_t)(kt + ty + r) * N + nt + tx];
  __syncthreads();
  #pragma unroll
  for (int r = 0; r < 32; r += 8)
    Wt[(size_t)(nt + ty + r) * K + kt + tx] = tile[tx][ty + r];
}

// ---------------- layernorm: fp32 row (C=1024) -> bf16 row ----------------
__global__ __launch_bounds__(128)
void ln_bf16_kernel(const float* __restrict__ x, const float* __restrict__ g,
                    const float* __restrict__ bt, bf16* __restrict__ out) {
  const int row  = blockIdx.x;
  const int wave = threadIdx.x >> 5, lane = threadIdx.x & 31;
  const float* xr = x + (size_t)row * 1024;
  float v[8]; float s = 0.f;
  #pragma unroll
  for (int i = 0; i < 8; ++i) { v[i] = xr[threadIdx.x + i * 128]; s += v[i]; }
  s = redsum32(s);
  __shared__ float red1[4], red2[4];
  if (lane == 0) red1[wave] = s;
  __syncthreads();
  const float mu = (red1[0] + red1[1] + red1[2] + red1[3]) * (1.f / 1024.f);
  float s2 = 0.f;
  #pragma unroll
  for (int i = 0; i < 8; ++i) { float d = v[i] - mu; s2 += d * d; }
  s2 = redsum32(s2);
  if (lane == 0) red2[wave] = s2;
  __syncthreads();
  const float var = (red2[0] + red2[1] + red2[2] + red2[3]) * (1.f / 1024.f);
  const float inv = rsqrtf(var + 1e-5f);
  #pragma unroll
  for (int i = 0; i < 8; ++i) {
    int c = threadIdx.x + i * 128;
    out[(size_t)row * 1024 + c] = (bf16)((v[i] - mu) * inv * g[c] + bt[c]);
  }
}

// ---------------- generic WMMA GEMM: C = A[M][K](bf16) x Bt[N][K](bf16) + bias -------
// block = 128 threads (4 waves, 2x2), block tile 128x128, wave tile 64x64 (4x4 frags)
#define GEMM_MODE_QK    0  // store bf16 into [B*H][T][64]  (head layout)
#define GEMM_MODE_VT    1  // store bf16 into [B*H][64][T]  (transposed head layout)
#define GEMM_MODE_GELU  2  // store bf16 gelu(val) into [M][N]
#define GEMM_MODE_RESID 3  // store f32 val + resid[m][n] into [M][N]

__global__ __launch_bounds__(128, 1)
void gemm_bf16_kernel(const bf16* __restrict__ A, const bf16* __restrict__ Bt,
                      const float* __restrict__ bias, void* __restrict__ out,
                      const float* __restrict__ resid,
                      int M, int N, int K, int mode) {
  const int lane = threadIdx.x & 31, wave = threadIdx.x >> 5;
  const int mWave = blockIdx.y * 128 + (wave >> 1) * 64;
  const int nWave = blockIdx.x * 128 + (wave & 1) * 64;
  const int half = lane >> 4, l16 = lane & 15;

  // per-lane row pointers, bumped by 32 elements per K step
  const bf16* pa0 = A + (size_t)(mWave +  0 + l16) * K + half * 8;
  const bf16* pa1 = A + (size_t)(mWave + 16 + l16) * K + half * 8;
  const bf16* pa2 = A + (size_t)(mWave + 32 + l16) * K + half * 8;
  const bf16* pa3 = A + (size_t)(mWave + 48 + l16) * K + half * 8;
  const bf16* pb0 = Bt + (size_t)(nWave +  0 + l16) * K + half * 8;
  const bf16* pb1 = Bt + (size_t)(nWave + 16 + l16) * K + half * 8;
  const bf16* pb2 = Bt + (size_t)(nWave + 32 + l16) * K + half * 8;
  const bf16* pb3 = Bt + (size_t)(nWave + 48 + l16) * K + half * 8;

  v8f acc[4][4] = {};
  for (int kb = 0; kb < K; kb += 32) {
    v16bf b0 = load_b_at(pb0); pb0 += 32;
    v16bf b1 = load_b_at(pb1); pb1 += 32;
    v16bf b2 = load_b_at(pb2); pb2 += 32;
    v16bf b3 = load_b_at(pb3); pb3 += 32;
    {
      v16bf a = load_a_at(pa0); pa0 += 32;
      acc[0][0] = wmma_bf16(a, b0, acc[0][0]);
      acc[0][1] = wmma_bf16(a, b1, acc[0][1]);
      acc[0][2] = wmma_bf16(a, b2, acc[0][2]);
      acc[0][3] = wmma_bf16(a, b3, acc[0][3]);
    }
    {
      v16bf a = load_a_at(pa1); pa1 += 32;
      acc[1][0] = wmma_bf16(a, b0, acc[1][0]);
      acc[1][1] = wmma_bf16(a, b1, acc[1][1]);
      acc[1][2] = wmma_bf16(a, b2, acc[1][2]);
      acc[1][3] = wmma_bf16(a, b3, acc[1][3]);
    }
    {
      v16bf a = load_a_at(pa2); pa2 += 32;
      acc[2][0] = wmma_bf16(a, b0, acc[2][0]);
      acc[2][1] = wmma_bf16(a, b1, acc[2][1]);
      acc[2][2] = wmma_bf16(a, b2, acc[2][2]);
      acc[2][3] = wmma_bf16(a, b3, acc[2][3]);
    }
    {
      v16bf a = load_a_at(pa3); pa3 += 32;
      acc[3][0] = wmma_bf16(a, b0, acc[3][0]);
      acc[3][1] = wmma_bf16(a, b1, acc[3][1]);
      acc[3][2] = wmma_bf16(a, b2, acc[3][2]);
      acc[3][3] = wmma_bf16(a, b3, acc[3][3]);
    }
  }

  #pragma unroll
  for (int i = 0; i < 4; ++i) {
    #pragma unroll
    for (int j = 0; j < 4; ++j) {
      #pragma unroll
      for (int r = 0; r < 8; ++r) {
        const int m = mWave + 16 * i + r + 8 * half;
        const int n = nWave + 16 * j + l16;
        float val = acc[i][j][r] + bias[n];
        if (mode == GEMM_MODE_QK) {
          int bb = m >> 11, t = m & 2047, h = n >> 6, d = n & 63;
          ((bf16*)out)[(((size_t)(bb * 16 + h) * 2048 + t) << 6) + d] = (bf16)val;
        } else if (mode == GEMM_MODE_VT) {
          int bb = m >> 11, t = m & 2047, h = n >> 6, d = n & 63;
          ((bf16*)out)[((size_t)(bb * 16 + h) * 64 + d) * 2048 + t] = (bf16)val;
        } else if (mode == GEMM_MODE_GELU) {
          float gl = 0.5f * val * (1.f + erff(val * 0.70710678118654752f));
          ((bf16*)out)[(size_t)m * N + n] = (bf16)gl;
        } else {
          ((float*)out)[(size_t)m * N + n] = val + resid[(size_t)m * N + n];
        }
      }
    }
  }
}

// ---------------- flash attention with hybrid mask ----------------
// q,k: [B*H][T][64] bf16 ; vT: [B*H][64][T] bf16 ; out: [B*T][1024] bf16
// grid = (T/64, B*H), block = 128 (4 waves x 16 query rows)
__global__ __launch_bounds__(128, 1)
void attn_kernel(const bf16* __restrict__ q, const bf16* __restrict__ k,
                 const bf16* __restrict__ vT, bf16* __restrict__ out,
                 const int* __restrict__ cond_ptr) {
  const int bh   = blockIdx.y;
  const int bb   = bh >> 4, h = bh & 15;
  const int wave = threadIdx.x >> 5, lane = threadIdx.x & 31;
  const int qBase = blockIdx.x * 64 + wave * 16;
  const int cond  = *cond_ptr;
  const int half = lane >> 4, col0 = lane & 15;

  const bf16* qh = q  + (size_t)bh * 2048 * 64;
  const bf16* kh = k  + (size_t)bh * 2048 * 64;
  const bf16* vh = vT + (size_t)bh * 64 * 2048;

  const v16bf qa0 = load_a_frag(qh, 64, qBase, 0, lane);
  const v16bf qa1 = load_a_frag(qh, 64, qBase, 32, lane);

  v8f yacc[4] = {};
  float mrow[8], lrow[8];
  #pragma unroll
  for (int r = 0; r < 8; ++r) { mrow[r] = -__builtin_inff(); lrow[r] = 0.f; }

  __shared__ bf16 ptile[4][16 * 64];
  bf16* myP = ptile[wave];

  const int jEnd = (cond > qBase + 16) ? cond : (qBase + 16);
  for (int jb = 0; jb < jEnd; jb += 64) {
    // ---- S = q @ k^T (K-dim = head_dim 64) ----
    v8f s[4];
    #pragma unroll
    for (int nf = 0; nf < 4; ++nf) {
      v8f z = {};
      z = wmma_bf16(qa0, load_b_frag(kh, 64, jb + nf * 16, 0, lane), z);
      z = wmma_bf16(qa1, load_b_frag(kh, 64, jb + nf * 16, 32, lane), z);
      s[nf] = z;
    }
    // ---- scale + hybrid mask + row max ----
    float nmax[8];
    #pragma unroll
    for (int r = 0; r < 8; ++r) nmax[r] = -__builtin_inff();
    #pragma unroll
    for (int nf = 0; nf < 4; ++nf) {
      const int j = jb + nf * 16 + col0;
      #pragma unroll
      for (int r = 0; r < 8; ++r) {
        const int i = qBase + r + 8 * half;
        const bool ok = (j < cond) || (i >= cond && j >= cond && j <= i);
        float sv = ok ? s[nf][r] * 0.125f : -__builtin_inff();
        s[nf][r] = sv;
        nmax[r] = fmaxf(nmax[r], sv);
      }
    }
    #pragma unroll
    for (int r = 0; r < 8; ++r) nmax[r] = redmax16(nmax[r]);
    // ---- online softmax update ----
    float corr[8], rsum[8];
    #pragma unroll
    for (int r = 0; r < 8; ++r) {
      float mn = fmaxf(mrow[r], nmax[r]);
      corr[r] = __expf(mrow[r] - mn);
      mrow[r] = mn;
      rsum[r] = 0.f;
    }
    #pragma unroll
    for (int nf = 0; nf < 4; ++nf)
      #pragma unroll
      for (int r = 0; r < 8; ++r) {
        float p = __expf(s[nf][r] - mrow[r]);
        s[nf][r] = p;
        rsum[r] += p;
      }
    #pragma unroll
    for (int r = 0; r < 8; ++r) {
      rsum[r] = redsum16(rsum[r]);
      lrow[r] = lrow[r] * corr[r] + rsum[r];
    }
    #pragma unroll
    for (int nf = 0; nf < 4; ++nf)
      #pragma unroll
      for (int r = 0; r < 8; ++r) yacc[nf][r] *= corr[r];
    // ---- D-layout -> A-layout via LDS ----
    #pragma unroll
    for (int nf = 0; nf < 4; ++nf)
      #pragma unroll
      for (int r = 0; r < 8; ++r)
        myP[(r + 8 * half) * 64 + nf * 16 + col0] = (bf16)s[nf][r];
    const v16bf pa0 = load_a_frag(myP, 64, 0, 0, lane);
    const v16bf pa1 = load_a_frag(myP, 64, 0, 32, lane);
    // ---- y += P @ V (K-dim = key block 64; B from vT[d][t]) ----
    #pragma unroll
    for (int nf = 0; nf < 4; ++nf) {
      yacc[nf] = wmma_bf16(pa0, load_b_frag(vh, 2048, nf * 16, jb, lane), yacc[nf]);
      yacc[nf] = wmma_bf16(pa1, load_b_frag(vh, 2048, nf * 16, jb + 32, lane), yacc[nf]);
    }
  }
  // ---- epilogue: normalize and store to [B*T][1024] ----
  #pragma unroll
  for (int nf = 0; nf < 4; ++nf)
    #pragma unroll
    for (int r = 0; r < 8; ++r) {
      const int t = qBase + r + 8 * half;
      const int d = nf * 16 + col0;
      const float val = yacc[nf][r] / lrow[r];
      out[((size_t)(bb * 2048 + t)) * 1024 + h * 64 + d] = (bf16)val;
    }
}

// ---------------- host launcher ----------------
extern "C" void kernel_launch(void* const* d_in, const int* in_sizes, int n_in,
                              void* d_out, int out_size, void* d_ws, size_t ws_size,
                              hipStream_t stream) {
  (void)in_sizes; (void)n_in; (void)out_size; (void)ws_size;
  const float* x     = (const float*)d_in[0];
  const float* ln1_g = (const float*)d_in[1];
  const float* ln1_b = (const float*)d_in[2];
  const float* Wk    = (const float*)d_in[3];
  const float* bk    = (const float*)d_in[4];
  const float* Wq    = (const float*)d_in[5];
  const float* bq    = (const float*)d_in[6];
  const float* Wv    = (const float*)d_in[7];
  const float* bv    = (const float*)d_in[8];
  const float* Wp    = (const float*)d_in[9];
  const float* bp    = (const float*)d_in[10];
  const float* ln2_g = (const float*)d_in[11];
  const float* ln2_b = (const float*)d_in[12];
  const float* W1    = (const float*)d_in[13];
  const float* b1    = (const float*)d_in[14];
  const float* W2    = (const float*)d_in[15];
  const float* b2    = (const float*)d_in[16];
  const int*   cond  = (const int*)d_in[17];

  char* ws = (char*)d_ws;
  const size_t MB = 1024ull * 1024ull;
  // arena with lifetime-based reuse (total 136 MB)
  bf16*  wkT  = (bf16*)(ws + 0 * MB);
  bf16*  wqT  = (bf16*)(ws + 2 * MB);
  bf16*  wvT  = (bf16*)(ws + 4 * MB);
  bf16*  wpT  = (bf16*)(ws + 6 * MB);
  bf16*  w1T  = (bf16*)(ws + 8 * MB);
  bf16*  w2T  = (bf16*)(ws + 16 * MB);
  bf16*  hln  = (bf16*)(ws + 24 * MB);   // ln1 out; dead after QKV
  bf16*  qb   = (bf16*)(ws + 40 * MB);
  bf16*  kb   = (bf16*)(ws + 56 * MB);
  bf16*  vT   = (bf16*)(ws + 72 * MB);
  bf16*  attO = (bf16*)(ws + 24 * MB);   // reuse hln; dead after proj
  float* x2   = (float*)(ws + 104 * MB); // proj residual out; live till end
  bf16*  h2   = (bf16*)(ws + 24 * MB);   // reuse attO region after proj
  bf16*  m1   = (bf16*)(ws + 40 * MB);   // reuse q/k/vT region (64 MB)

  const int M = 8192;  // B*T

  // 1) weight transpose + cast (LDS tiled)
  transpose_bf16_kernel<<<dim3(32, 32),  dim3(32, 8), 0, stream>>>(Wk, wkT, 1024, 1024);
  transpose_bf16_kernel<<<dim3(32, 32),  dim3(32, 8), 0, stream>>>(Wq, wqT, 1024, 1024);
  transpose_bf16_kernel<<<dim3(32, 32),  dim3(32, 8), 0, stream>>>(Wv, wvT, 1024, 1024);
  transpose_bf16_kernel<<<dim3(32, 32),  dim3(32, 8), 0, stream>>>(Wp, wpT, 1024, 1024);
  transpose_bf16_kernel<<<dim3(128, 32), dim3(32, 8), 0, stream>>>(W1, w1T, 1024, 4096);
  transpose_bf16_kernel<<<dim3(32, 128), dim3(32, 8), 0, stream>>>(W2, w2T, 4096, 1024);

  // 2) LN1
  ln_bf16_kernel<<<M, 128, 0, stream>>>(x, ln1_g, ln1_b, hln);

  // 3) QKV projections (head layouts)
  gemm_bf16_kernel<<<dim3(8, 64), 128, 0, stream>>>(hln, wqT, bq, qb, nullptr, M, 1024, 1024, GEMM_MODE_QK);
  gemm_bf16_kernel<<<dim3(8, 64), 128, 0, stream>>>(hln, wkT, bk, kb, nullptr, M, 1024, 1024, GEMM_MODE_QK);
  gemm_bf16_kernel<<<dim3(8, 64), 128, 0, stream>>>(hln, wvT, bv, vT, nullptr, M, 1024, 1024, GEMM_MODE_VT);

  // 4) attention (hybrid mask, online softmax)
  attn_kernel<<<dim3(32, 64), 128, 0, stream>>>(qb, kb, vT, attO, cond);

  // 5) output projection + residual -> x2 (fp32)
  gemm_bf16_kernel<<<dim3(8, 64), 128, 0, stream>>>(attO, wpT, bp, x2, x, M, 1024, 1024, GEMM_MODE_RESID);

  // 6) LN2
  ln_bf16_kernel<<<M, 128, 0, stream>>>(x2, ln2_g, ln2_b, h2);

  // 7) MLP: gelu(h2 @ W1 + b1) @ W2 + b2 + x2 -> d_out (fp32)
  gemm_bf16_kernel<<<dim3(32, 64), 128, 0, stream>>>(h2, w1T, b1, m1, nullptr, M, 4096, 1024, GEMM_MODE_GELU);
  gemm_bf16_kernel<<<dim3(8, 64), 128, 0, stream>>>(m1, w2T, b2, (float*)d_out, x2, M, 1024, 4096, GEMM_MODE_RESID);
}
